// KAN_15710990369708
// MI455X (gfx1250) — matrix operand order repaired and checked by reading, not compiled, not run on hardware
//
#include <hip/hip_runtime.h>
#include <hip/hip_bf16.h>
#include <math.h>

// ---------------------------------------------------------------------------
// KAN forward, 2 layers, B=4096, DIN=DOUT=1024, COEFFS=11.
// Per layer: A = [silu(x) | 11 bspline bases] bf16 (K=12288),
//            W = [base_w | spline_w*scaler]   bf16,
// then one bf16 WMMA GEMM (f32 accum): out = A @ W^T.
// GEMM uses double-buffered GLOBAL_LOAD_ASYNC_TO_LDS_B128 staging (ASYNCcnt).
// ---------------------------------------------------------------------------

typedef __attribute__((ext_vector_type(16))) __bf16 v16bf;
typedef __attribute__((ext_vector_type(8)))  __bf16 v8bf;
typedef __attribute__((ext_vector_type(8)))  float  v8f;
typedef int v4i __attribute__((vector_size(16)));

#define B_ROWS   4096
#define DIN      1024
#define DOUT     1024
#define COEFFS   11
#define KTOT     (12 * DIN)      // 12288
#define NGRID    15
#define BK       64              // K per pipeline stage (2 WMMA sub-steps)
#define LDSTR    72              // 64 + 8 bf16 pad per LDS row (144B, 16B-mult)

#if defined(__AMDGCN__) && __has_builtin(__builtin_amdgcn_global_load_async_to_lds_b128)
#define USE_ASYNC 1
#else
#define USE_ASYNC 0
#endif

#if USE_ASYNC
// builtin prototype (probe-derived): (AS1 v4i*, AS3 v4i*, imm offset, imm cpol)
typedef __attribute__((address_space(1))) v4i* gptr_v4i;
typedef __attribute__((address_space(3))) v4i* lptr_v4i;
#if __has_builtin(__builtin_amdgcn_s_wait_asynccnt)
#define WAIT_ASYNC(n) __builtin_amdgcn_s_wait_asynccnt(n)
#else
#define WAIT_ASYNC(n) asm volatile("s_wait_asynccnt %0" ::"n"(n))
#endif
#endif

// round-to-nearest-even f32 -> bf16 bits
__device__ __forceinline__ unsigned short f2bf(float f) {
    union { float f; unsigned u; } v; v.f = f;
    unsigned r = (v.u + 0x7FFFu + ((v.u >> 16) & 1u)) >> 16;
    return (unsigned short)r;
}

// ---------------------------------------------------------------------------
// Kernel 1: silu + Cox-de-Boor bases -> 12 bf16 per (b,i)
// ---------------------------------------------------------------------------
__global__ __launch_bounds__(256) void kan_prep_act(
        const float* __restrict__ xin,
        const float* __restrict__ grid,
        unsigned short* __restrict__ actA) {
    const long long idx = (long long)blockIdx.x * 256 + threadIdx.x;  // b*DIN+i
    const float x = xin[idx];
    const float s = x / (1.0f + __expf(-x));

    float g[NGRID];
#pragma unroll
    for (int j = 0; j < NGRID; ++j) g[j] = grid[j];

    float bas[NGRID - 1];
#pragma unroll
    for (int j = 0; j < NGRID - 1; ++j)
        bas[j] = (s >= g[j] && s < g[j + 1]) ? 1.0f : 0.0f;

#pragma unroll
    for (int k = 1; k <= 3; ++k) {
#pragma unroll
        for (int j = 0; j < NGRID - 1 - k; ++j) {
            float left  = (s - g[j]) / (g[j + k] - g[j]);
            float right = (g[j + k + 1] - s) / (g[j + k + 1] - g[j + 1]);
            bas[j] = left * bas[j] + right * bas[j + 1];
        }
    }

    union { unsigned short h[12]; uint2 q[3]; } u;
    u.h[0] = f2bf(s);
#pragma unroll
    for (int c = 0; c < COEFFS; ++c) u.h[1 + c] = f2bf(bas[c]);

    uint2* dst = (uint2*)(actA + idx * 12);
    dst[0] = u.q[0]; dst[1] = u.q[1]; dst[2] = u.q[2];
}

// ---------------------------------------------------------------------------
// Kernel 2: Wc[o, i*12+{0,1+c}] = {base_w[o,i], spline_w[o,i,c]*scaler[o,i]}
// ---------------------------------------------------------------------------
__global__ __launch_bounds__(256) void kan_prep_w(
        const float* __restrict__ base_w,
        const float* __restrict__ spline_w,
        const float* __restrict__ scaler,
        unsigned short* __restrict__ Wc) {
    const long long idx = (long long)blockIdx.x * 256 + threadIdx.x;  // o*DIN+i
    const float bw = base_w[idx];
    const float sc = scaler[idx];
    const float* sp = spline_w + idx * COEFFS;

    union { unsigned short h[12]; uint2 q[3]; } u;
    u.h[0] = f2bf(bw);
#pragma unroll
    for (int c = 0; c < COEFFS; ++c) u.h[1 + c] = f2bf(sp[c] * sc);

    uint2* dst = (uint2*)(Wc + idx * 12);
    dst[0] = u.q[0]; dst[1] = u.q[1]; dst[2] = u.q[2];
}

// ---------------------------------------------------------------------------
// Kernel 3: bf16 WMMA GEMM, C[4096,1024] = A[M,K] * W[N,K]^T.
// Block tile 128x128, 8 wave32s (2M x 4N), wave tile 64x32.
// Double-buffered async staging, BK=64 per stage.
// ---------------------------------------------------------------------------
__global__ __launch_bounds__(256) void kan_gemm(
        const unsigned short* __restrict__ A,
        const unsigned short* __restrict__ W,
        float* __restrict__ C) {
    __shared__ unsigned short ldsA[2 * 128 * LDSTR];
    __shared__ unsigned short ldsB[2 * 128 * LDSTR];

    const int tid  = threadIdx.x;
    const int lane = tid & 31;
    const int wave = tid >> 5;
    const int wm   = wave & 1;        // M direction (64 rows each)
    const int wn   = wave >> 1;       // N direction (32 cols each)
    const int half = lane >> 4;
    const int l16  = lane & 15;

    const int tile_m = blockIdx.y * 128;
    const int tile_n = blockIdx.x * 128;

    // staging: per stage each tile is 128 rows x 128B = 1024 b128 chunks;
    // 256 threads x 4 chunks, chunk c -> row = c>>3, seg = c&7 (16B units)
    const int c0row = tid >> 3;       // base row for j=0 (row += 32 per j)
    const int c0seg = tid & 7;

    const unsigned short* gA0 = A + (size_t)(tile_m + c0row) * KTOT + c0seg * 8;
    const unsigned short* gB0 = W + (size_t)(tile_n + c0row) * KTOT + c0seg * 8;
    const int ldsOff0 = c0row * LDSTR + c0seg * 8;

    auto stage = [&](int k0, int buf) {
        const int lbase = buf * 128 * LDSTR + ldsOff0;
#pragma unroll
        for (int j = 0; j < 4; ++j) {
            const unsigned short* ga = gA0 + (size_t)(32 * j) * KTOT + k0;
            const unsigned short* gb = gB0 + (size_t)(32 * j) * KTOT + k0;
            unsigned short* la = &ldsA[lbase + 32 * j * LDSTR];
            unsigned short* lb = &ldsB[lbase + 32 * j * LDSTR];
#if USE_ASYNC
            __builtin_amdgcn_global_load_async_to_lds_b128(
                (gptr_v4i)ga, (lptr_v4i)la, 0, 0);
            __builtin_amdgcn_global_load_async_to_lds_b128(
                (gptr_v4i)gb, (lptr_v4i)lb, 0, 0);
#else
            *(uint4*)la = *(const uint4*)ga;
            *(uint4*)lb = *(const uint4*)gb;
#endif
        }
    };

    v8f acc[4][2];
#pragma unroll
    for (int mt = 0; mt < 4; ++mt)
#pragma unroll
        for (int nt = 0; nt < 2; ++nt)
            acc[mt][nt] = (v8f)(0.0f);

    const int NSTEP = KTOT / BK;      // 192
    stage(0, 0);

    for (int step = 0; step < NSTEP; ++step) {
        const int cur = step & 1;
        const bool more = (step + 1) < NSTEP;
        if (more) stage((step + 1) * BK, cur ^ 1);
#if USE_ASYNC
        if (more) { WAIT_ASYNC(8); } else { WAIT_ASYNC(0); }
#endif
        __syncthreads();

        const unsigned short* bufA = ldsA + cur * 128 * LDSTR;
        const unsigned short* bufB = ldsB + cur * 128 * LDSTR;

#pragma unroll
        for (int ks = 0; ks < 2; ++ks) {          // two K=32 sub-steps
            const int ko = ks * 32;

            // A fragments (16x32 bf16, ISA 7.12.2):
            // lane m=l16; half 0 -> K={0..7,16..23}, half 1 -> K={8..15,24..31}
            v16bf af[4];
#pragma unroll
            for (int mt = 0; mt < 4; ++mt) {
                const unsigned short* p =
                    bufA + (wm * 64 + mt * 16 + l16) * LDSTR + ko;
                v8bf lo = *(const v8bf*)(p + 8 * half);
                v8bf hi = *(const v8bf*)(p + 16 + 8 * half);
                af[mt] = __builtin_shufflevector(lo, hi,
                    0, 1, 2, 3, 4, 5, 6, 7, 8, 9, 10, 11, 12, 13, 14, 15);
            }

            // B fragments (32x16 bf16): col n=l16; half 0 -> K={0..7,16..23}, half 1 -> K={8..15,24..31}
            v16bf bfr[2];
#pragma unroll
            for (int nt = 0; nt < 2; ++nt) {
                const unsigned short* p =
                    bufB + (wn * 32 + nt * 16 + l16) * LDSTR + ko;
                v8bf lo = *(const v8bf*)(p + 8 * half);
                v8bf hi = *(const v8bf*)(p + 16 + 8 * half);
                bfr[nt] = __builtin_shufflevector(lo, hi,
                    0, 1, 2, 3, 4, 5, 6, 7, 8, 9, 10, 11, 12, 13, 14, 15);
            }

#pragma unroll
            for (int mt = 0; mt < 4; ++mt)
#pragma unroll
                for (int nt = 0; nt < 2; ++nt)
                    acc[mt][nt] = __builtin_amdgcn_wmma_f32_16x16x32_bf16(
                        false, af[mt], false, bfr[nt],
                        (short)0, acc[mt][nt], false, false);
        }
        __syncthreads();
    }

    // C/D layout: VGPR r -> M = r + 8*half, N = l16
#pragma unroll
    for (int mt = 0; mt < 4; ++mt) {
#pragma unroll
        for (int nt = 0; nt < 2; ++nt) {
            const int gm0 = tile_m + wm * 64 + mt * 16 + 8 * half;
            const int gn  = tile_n + wn * 32 + nt * 16 + l16;
#pragma unroll
            for (int r = 0; r < 8; ++r)
                C[(size_t)(gm0 + r) * DOUT + gn] = acc[mt][nt][r];
        }
    }
}

// ---------------------------------------------------------------------------
// Host side
// ---------------------------------------------------------------------------
extern "C" void kernel_launch(void* const* d_in, const int* in_sizes, int n_in,
                              void* d_out, int out_size, void* d_ws, size_t ws_size,
                              hipStream_t stream) {
    const float* x        = (const float*)d_in[0];
    const float* base_w   = (const float*)d_in[1];   // [2,1024,1024]
    const float* spline_w = (const float*)d_in[2];   // [2,1024,1024,11]
    const float* scaler   = (const float*)d_in[3];   // [2,1024,1024]
    const float* grid     = (const float*)d_in[4];   // [15]

    char* ws = (char*)d_ws;
    const size_t actA_bytes = (size_t)B_ROWS * KTOT * 2;  // 96 MB
    const size_t Wc_bytes   = (size_t)DOUT  * KTOT * 2;   // 24 MB
    unsigned short* actA = (unsigned short*)ws;
    unsigned short* Wc   = (unsigned short*)(ws + actA_bytes);
    float* xbuf          = (float*)(ws + actA_bytes + Wc_bytes);  // 16 MB

    float* out = (float*)d_out;
    const size_t layer_w  = (size_t)DOUT * DIN;
    const size_t layer_sp = layer_w * COEFFS;

    for (int l = 0; l < 2; ++l) {
        const float* xin = (l == 0) ? x : xbuf;
        kan_prep_act<<<(B_ROWS * DIN) / 256, 256, 0, stream>>>(xin, grid, actA);
        kan_prep_w<<<(DOUT * DIN) / 256, 256, 0, stream>>>(
            base_w + l * layer_w, spline_w + l * layer_sp, scaler + l * layer_w, Wc);
        kan_gemm<<<dim3(DOUT / 128, B_ROWS / 128), 256, 0, stream>>>(
            actA, Wc, (l == 0) ? xbuf : out);
    }
}
